// TransformerEncoder_6863357739463
// MI455X (gfx1250) — matrix operand order repaired and checked by reading, not compiled, not run on hardware
//
#include <hip/hip_runtime.h>
#include <cstdint>
#include <cstddef>

#define B_SZ    16
#define S_LEN   1024
#define D_MODEL 512
#define N_HEADS 8
#define D_HEAD  64
#define N_LAYERS 6
#define FFN_DIM 2048
#define N_TOK   (B_SZ * S_LEN)

typedef __attribute__((ext_vector_type(16))) __bf16 v16bf;
typedef __attribute__((ext_vector_type(8)))  float  v8f;
typedef __attribute__((ext_vector_type(4)))  unsigned int v4u;
typedef __attribute__((ext_vector_type(8)))  int v8i;
typedef __attribute__((ext_vector_type(4)))  int v4i;

// Tensor Data Mover availability (device pass only; host pass falls back)
#if defined(__has_builtin)
#if __has_builtin(__builtin_amdgcn_tensor_load_to_lds) && __has_builtin(__builtin_amdgcn_s_wait_tensorcnt)
#define HAVE_TDM 1
#endif
#endif
#ifndef HAVE_TDM
#define HAVE_TDM 0
#endif

union FragBF {
  v16bf v;
  uint4 q[2];
};

__device__ __forceinline__ unsigned short f2bf(float f) {
  unsigned int u = __float_as_uint(f);
  u += 0x7FFFu + ((u >> 16) & 1u);          // round-to-nearest-even
  return (unsigned short)(u >> 16);
}

__device__ __forceinline__ v8f wmma_bf16(const FragBF& a, const FragBF& b, v8f c) {
  return __builtin_amdgcn_wmma_f32_16x16x32_bf16(false, a.v, false, b.v,
                                                 (short)0, c, false, false);
}

__device__ __forceinline__ v8f v8f_zero() {
  v8f z;
#pragma unroll
  for (int i = 0; i < 8; ++i) z[i] = 0.0f;
  return z;
}

// ---------------------------------------------------------------------------
// f32 -> bf16 conversion (4 elems / thread)
// ---------------------------------------------------------------------------
__global__ __launch_bounds__(256)
void cvt_bf16_kernel(const float* __restrict__ in, unsigned short* __restrict__ out, int n4) {
  int i = blockIdx.x * 256 + threadIdx.x;
  if (i >= n4) return;
  float4 v = ((const float4*)in)[i];
  unsigned int lo = (unsigned int)f2bf(v.x) | ((unsigned int)f2bf(v.y) << 16);
  unsigned int hi = (unsigned int)f2bf(v.z) | ((unsigned int)f2bf(v.w) << 16);
  ((uint2*)out)[i] = make_uint2(lo, hi);
}

// ---------------------------------------------------------------------------
// Embedding + positional encoding + input mask
// ---------------------------------------------------------------------------
__global__ __launch_bounds__(256)
void embed_kernel(const int* __restrict__ inp, const int* __restrict__ mask,
                  const float* __restrict__ emb, const float* __restrict__ pos,
                  float* __restrict__ x) {
  int idx = blockIdx.x * 256 + threadIdx.x;     // N_TOK * 128 threads total
  int tok = idx >> 7;
  int c   = (idx & 127) * 4;
  int s   = tok & (S_LEN - 1);
  int id  = inp[tok];
  float mf = (float)mask[tok];
  float4 e = *(const float4*)&emb[(size_t)id * D_MODEL + c];
  float4 p = *(const float4*)&pos[(size_t)s  * D_MODEL + c];
  float4 r;
  r.x = (e.x + p.x) * mf; r.y = (e.y + p.y) * mf;
  r.z = (e.z + p.z) * mf; r.w = (e.w + p.w) * mf;
  *(float4*)&x[(size_t)tok * D_MODEL + c] = r;
}

// ---------------------------------------------------------------------------
// bf16 GEMM:  C[M,N] = A[M,K] * B[N,K]^T (+bias)(+residual)(relu)
// Block = 256 threads (8 waves), tile 128x128x32, wave grid 2x4, each wave
// owns a 64x32 sub-tile = 4x2 WMMA 16x16 f32 accumulators.
// LDS rows padded to 40 halfwords (20 banks) -> conflict-free b128 frag reads.
// B (weight) tile is moved by the Tensor Data Mover with hardware pad
// insertion (64B row + 16B pad); A tile uses vector loads + prefetch.
// ---------------------------------------------------------------------------
template<bool RELU, bool OUT_BF16>
__global__ __launch_bounds__(256)
void gemm_bt_kernel(const unsigned short* __restrict__ A,
                    const unsigned short* __restrict__ Bw,
                    const float* __restrict__ bias,
                    const float* __restrict__ residual,
                    void* __restrict__ Cout,
                    int M, int N, int K) {
  constexpr int LDS_S = 40;
  __shared__ unsigned short As[128 * LDS_S];
  __shared__ unsigned short Bs[128 * LDS_S];

  const int t    = threadIdx.x;
  const int lane = t & 31;
  const int wave = t >> 5;
  const int wm   = wave >> 2;         // 0..1
  const int wn   = wave & 3;          // 0..3
  const int half = lane >> 4;         // 0/1
  const int lr   = lane & 15;
  const int row0 = blockIdx.x * 128;
  const int col0 = blockIdx.y * 128;

  v8f acc[4][2];
#pragma unroll
  for (int i = 0; i < 4; ++i)
#pragma unroll
    for (int j = 0; j < 2; ++j) acc[i][j] = v8f_zero();

  for (int k0 = 0; k0 < K; k0 += 32) {
    __syncthreads();
    // A tile: 128x32 bf16, vector loads
#pragma unroll
    for (int i = 0; i < 2; ++i) {
      int idx = t + i * 256;
      int r = idx >> 2;
      int c = (idx & 3) * 8;
      *(uint4*)&As[r * LDS_S + c] = *(const uint4*)&A[(size_t)(row0 + r) * K + k0 + c];
      if (k0 + 32 < K)
        __builtin_prefetch(&A[(size_t)(row0 + r) * K + k0 + 32 + c], 0, 1);
    }
#if HAVE_TDM
    // B tile: 128 rows x 32 bf16 via TDM, LDS rows padded 64B+16B (stride 80B)
    if (wave == 0) {
      unsigned long long ga =
          (unsigned long long)(const void*)(Bw + (size_t)col0 * K + k0);
      unsigned int ldsb = (unsigned int)(size_t)(void*)Bs;   // LDS byte offset
      v4u g0 = { 1u,                                          // count=1
                 ldsb,                                        // lds_addr
                 (unsigned int)ga,                            // global_addr lo
                 (unsigned int)((ga >> 32) & 0x01FFFFFFull)   // global_addr hi (56:32)
                     | 0x80000000u };                         // type=2 (bits 127:126)
      unsigned int ku = (unsigned int)K;
      v8i g1 = { (int)(0x00010000u                            // data_size=1 (2B)
                       | (1u << 20)                           // pad_enable
                       | (3u << 22)                           // pad_interval: 16 DWORDs
                       | (3u << 25)),                         // pad_amount: 4 DWORDs
                 (int)((ku & 0xFFFFu) << 16),                 // tensor_dim0 lo16
                 (int)(((ku >> 16) & 0xFFFFu) | (128u << 16)),// dim0 hi16 | tensor_dim1=128
                 (int)(32u << 16),                            // dim1 hi=0 | tile_dim0=32
                 (int)(128u),                                 // tile_dim1=128 | tile_dim2=0
                 (int)ku,                                     // dim0_stride lo32 = K
                 0, 0 };                                      // stride hi / dim1_stride
      v4i g2 = {0, 0, 0, 0};
      v4i g3 = {0, 0, 0, 0};
      v8i g4 = {0, 0, 0, 0, 0, 0, 0, 0};
      __builtin_amdgcn_tensor_load_to_lds(g0, g1, g2, g3, g4, 0);
      __builtin_amdgcn_s_wait_tensorcnt(0);
    }
#else
#pragma unroll
    for (int i = 0; i < 2; ++i) {
      int idx = t + i * 256;
      int r = idx >> 2;
      int c = (idx & 3) * 8;
      *(uint4*)&Bs[r * LDS_S + c] = *(const uint4*)&Bw[(size_t)(col0 + r) * K + k0 + c];
    }
#endif
    __syncthreads();

    FragBF a[4], b[2];
#pragma unroll
    for (int ms = 0; ms < 4; ++ms) {
      // A 16x32 frag: elems 0..7 = K[half*8..+7], elems 8..15 = K[16+half*8..+7]
      int base = (wm * 64 + ms * 16 + lr) * LDS_S + half * 8;
      a[ms].q[0] = *(const uint4*)&As[base];
      a[ms].q[1] = *(const uint4*)&As[base + 16];
    }
#pragma unroll
    for (int ns = 0; ns < 2; ++ns) {
      // B 32x16 frag: lane holds column n=lr, elems 0..7 = K[half*8..+7], elems 8..15 = K[16+half*8..+7]
      int base = (wn * 32 + ns * 16 + lr) * LDS_S + half * 8;
      b[ns].q[0] = *(const uint4*)&Bs[base];
      b[ns].q[1] = *(const uint4*)&Bs[base + 16];
    }
#pragma unroll
    for (int ms = 0; ms < 4; ++ms)
#pragma unroll
      for (int ns = 0; ns < 2; ++ns)
        acc[ms][ns] = wmma_bf16(a[ms], b[ns], acc[ms][ns]);
  }

#pragma unroll
  for (int ms = 0; ms < 4; ++ms)
#pragma unroll
    for (int ns = 0; ns < 2; ++ns)
#pragma unroll
      for (int r = 0; r < 8; ++r) {
        int gr = row0 + wm * 64 + ms * 16 + r + 8 * half;   // C layout: M=r(+8), N=lane
        int gc = col0 + wn * 32 + ns * 16 + lr;
        float v = acc[ms][ns][r];
        if (bias)     v += bias[gc];
        if (residual) v += residual[(size_t)gr * N + gc];
        if (RELU)     v = fmaxf(v, 0.0f);
        if (OUT_BF16) ((unsigned short*)Cout)[(size_t)gr * N + gc] = f2bf(v);
        else          ((float*)Cout)[(size_t)gr * N + gc] = v;
      }
}

// ---------------------------------------------------------------------------
// Flash attention: one block = one (b, h, 128-query tile). 8 waves, each wave
// owns 16 query rows. K/V processed in tiles of 64 keys. All matmuls WMMA bf16.
// Masked scores get -1e20 BEFORE the 1/sqrt(dh) scale (matches reference).
// ---------------------------------------------------------------------------
__global__ __launch_bounds__(256)
void flash_attn_kernel(const unsigned short* __restrict__ qb,
                       const unsigned short* __restrict__ kb,
                       const unsigned short* __restrict__ vb,
                       const int* __restrict__ mask,
                       unsigned short* __restrict__ ob) {
  constexpr int QS = 72;  // padded LDS row stride (36 banks) -> conflict-free
  __shared__ unsigned short Qs[128 * QS];
  __shared__ unsigned short Ks[64 * QS];
  __shared__ unsigned short Vt[64 * QS];       // V transposed: [dh][key]
  __shared__ unsigned short Ps[8 * 16 * QS];   // per-wave P tile [16][64]

  const int t    = threadIdx.x;
  const int lane = t & 31;
  const int wave = t >> 5;
  const int half = lane >> 4;
  const int lr   = lane & 15;
  const int q0   = blockIdx.x * 128;
  const int bh   = blockIdx.y;
  const int b    = bh >> 3;
  const int h    = bh & 7;
  const size_t rowbase = (size_t)b * S_LEN;
  const int coloff = h * D_HEAD;

  // Q tile -> LDS
#pragma unroll
  for (int i = 0; i < 4; ++i) {
    int idx = t + i * 256;
    int r = idx >> 3;
    int c = (idx & 7) * 8;
    *(uint4*)&Qs[r * QS + c] =
        *(const uint4*)&qb[(rowbase + q0 + r) * D_MODEL + coloff + c];
  }

  v8f accO[4];
#pragma unroll
  for (int i = 0; i < 4; ++i) accO[i] = v8f_zero();
  float mrow[8], lrow[8];
#pragma unroll
  for (int r = 0; r < 8; ++r) { mrow[r] = -3.0e38f; lrow[r] = 0.0f; }

  for (int kv0 = 0; kv0 < S_LEN; kv0 += 64) {
    __syncthreads();
    // K tile (row-major) and V tile (transposed) -> LDS
#pragma unroll
    for (int i = 0; i < 2; ++i) {
      int idx = t + i * 256;
      int r = idx >> 3;
      int c = (idx & 7) * 8;
      *(uint4*)&Ks[r * QS + c] =
          *(const uint4*)&kb[(rowbase + kv0 + r) * D_MODEL + coloff + c];
      uint4 vv = *(const uint4*)&vb[(rowbase + kv0 + r) * D_MODEL + coloff + c];
      const unsigned short* pv = (const unsigned short*)&vv;
#pragma unroll
      for (int j = 0; j < 8; ++j) Vt[(c + j) * QS + r] = pv[j];
    }
    __syncthreads();

    // S = Q @ K^T : wave rows [wave*16, +16), 4 column tiles of 16 keys
    v8f accS[4];
#pragma unroll
    for (int i = 0; i < 4; ++i) accS[i] = v8f_zero();
#pragma unroll
    for (int kd = 0; kd < 64; kd += 32) {
      FragBF aq;
      int abase = (wave * 16 + lr) * QS + kd + half * 8;
      aq.q[0] = *(const uint4*)&Qs[abase];
      aq.q[1] = *(const uint4*)&Qs[abase + 16];
#pragma unroll
      for (int nt = 0; nt < 4; ++nt) {
        FragBF bk_;
        int bbase = (nt * 16 + lr) * QS + kd + half * 8;
        bk_.q[0] = *(const uint4*)&Ks[bbase];
        bk_.q[1] = *(const uint4*)&Ks[bbase + 16];
        accS[nt] = wmma_bf16(aq, bk_, accS[nt]);
      }
    }

    // mask-fill then scale
    int keep[4];
#pragma unroll
    for (int nt = 0; nt < 4; ++nt)
      keep[nt] = mask[b * S_LEN + kv0 + nt * 16 + lr];
#pragma unroll
    for (int nt = 0; nt < 4; ++nt)
#pragma unroll
      for (int r = 0; r < 8; ++r) {
        float s = accS[nt][r];
        accS[nt][r] = keep[nt] ? s * 0.125f : -1.25e19f;  // (-1e20)/8
      }

    // online softmax; row stats per accumulator slot, shared via 16-lane butterfly
#pragma unroll
    for (int r = 0; r < 8; ++r) {
      float mloc = accS[0][r];
#pragma unroll
      for (int nt = 1; nt < 4; ++nt) mloc = fmaxf(mloc, accS[nt][r]);
#pragma unroll
      for (int off = 1; off < 16; off <<= 1)
        mloc = fmaxf(mloc, __shfl_xor(mloc, off));
      float mnew = fmaxf(mrow[r], mloc);
      float rowsum = 0.0f;
#pragma unroll
      for (int nt = 0; nt < 4; ++nt) {
        float p = __expf(accS[nt][r] - mnew);
        accS[nt][r] = p;
        rowsum += p;
      }
#pragma unroll
      for (int off = 1; off < 16; off <<= 1)
        rowsum += __shfl_xor(rowsum, off);
      float corr = __expf(mrow[r] - mnew);
      lrow[r] = lrow[r] * corr + rowsum;
      mrow[r] = mnew;
#pragma unroll
      for (int j = 0; j < 4; ++j) accO[j][r] *= corr;
    }

    // P (C-layout f32) -> per-wave LDS region as bf16 (wave DS ops are in-order)
#pragma unroll
    for (int nt = 0; nt < 4; ++nt)
#pragma unroll
      for (int r = 0; r < 8; ++r)
        Ps[(wave * 16 + r + 8 * half) * QS + nt * 16 + lr] = f2bf(accS[nt][r]);

    // O += P @ V  (contraction over 64 keys, Vt gives contiguous B frags)
#pragma unroll
    for (int kd = 0; kd < 64; kd += 32) {
      FragBF ap;
      int abase = (wave * 16 + lr) * QS + kd + half * 8;
      ap.q[0] = *(const uint4*)&Ps[abase];
      ap.q[1] = *(const uint4*)&Ps[abase + 16];
#pragma unroll
      for (int nt = 0; nt < 4; ++nt) {
        FragBF bv;
        int bbase = (nt * 16 + lr) * QS + kd + half * 8;
        bv.q[0] = *(const uint4*)&Vt[bbase];
        bv.q[1] = *(const uint4*)&Vt[bbase + 16];
        accO[nt] = wmma_bf16(ap, bv, accO[nt]);
      }
    }
  }

  // normalize and write bf16
#pragma unroll
  for (int nt = 0; nt < 4; ++nt)
#pragma unroll
    for (int r = 0; r < 8; ++r) {
      int gr = q0 + wave * 16 + r + 8 * half;
      int gc = coloff + nt * 16 + lr;
      float v = accO[nt][r] / fmaxf(lrow[r], 1e-20f);
      ob[(rowbase + gr) * D_MODEL + gc] = f2bf(v);
    }
}

// ---------------------------------------------------------------------------
// LayerNorm over D=512, one wave per token; optional mask multiply at the end.
// ---------------------------------------------------------------------------
__global__ __launch_bounds__(256)
void layernorm_kernel(const float* __restrict__ in,
                      const float* __restrict__ gamma,
                      const float* __restrict__ beta,
                      const int* __restrict__ mask,   // nullable
                      float* __restrict__ out) {
  const int lane = threadIdx.x & 31;
  const int wave = threadIdx.x >> 5;
  const int tok  = blockIdx.x * 8 + wave;
  const float* x = in + (size_t)tok * D_MODEL;

  float4 v[4];
  float s = 0.0f, ss = 0.0f;
#pragma unroll
  for (int i = 0; i < 4; ++i) {
    v[i] = *(const float4*)&x[i * 128 + lane * 4];
    s  += v[i].x + v[i].y + v[i].z + v[i].w;
    ss += v[i].x * v[i].x + v[i].y * v[i].y + v[i].z * v[i].z + v[i].w * v[i].w;
  }
#pragma unroll
  for (int off = 1; off < 32; off <<= 1) {
    s  += __shfl_xor(s, off);
    ss += __shfl_xor(ss, off);
  }
  float mean = s * (1.0f / 512.0f);
  float var  = ss * (1.0f / 512.0f) - mean * mean;
  float rstd = rsqrtf(var + 1e-5f);
  float mf = mask ? (float)mask[tok] : 1.0f;
  float* o = out + (size_t)tok * D_MODEL;
#pragma unroll
  for (int i = 0; i < 4; ++i) {
    int d = i * 128 + lane * 4;
    float4 g  = *(const float4*)&gamma[d];
    float4 bb = *(const float4*)&beta[d];
    float4 r;
    r.x = ((v[i].x - mean) * rstd * g.x + bb.x) * mf;
    r.y = ((v[i].y - mean) * rstd * g.y + bb.y) * mf;
    r.z = ((v[i].z - mean) * rstd * g.z + bb.z) * mf;
    r.w = ((v[i].w - mean) * rstd * g.w + bb.w) * mf;
    *(float4*)&o[d] = r;
  }
}

// ---------------------------------------------------------------------------
// Masked mean pool over sequence: out[b,d] = sum_s x[b,s,d] / max(sum_s m, 1e-20)
// ---------------------------------------------------------------------------
__global__ __launch_bounds__(256)
void pool_kernel(const float* __restrict__ x, const int* __restrict__ mask,
                 float* __restrict__ out) {
  __shared__ float red[256];
  const int b = blockIdx.x;
  const int t = threadIdx.x;
  float cnt = 0.0f;
  for (int s = t; s < S_LEN; s += 256) cnt += (float)mask[b * S_LEN + s];
  red[t] = cnt;
  __syncthreads();
  for (int w = 128; w > 0; w >>= 1) {
    if (t < w) red[t] += red[t + w];
    __syncthreads();
  }
  float div = fmaxf(red[0], 1e-20f);

  float a0 = 0.0f, a1 = 0.0f;
  for (int s = 0; s < S_LEN; ++s) {
    const float* row = x + ((size_t)(b * S_LEN + s)) * D_MODEL;
    a0 += row[t];
    a1 += row[t + 256];
  }
  out[b * D_MODEL + t]       = a0 / div;
  out[b * D_MODEL + t + 256] = a1 / div;
}

// ---------------------------------------------------------------------------
// Host orchestration
// ---------------------------------------------------------------------------
extern "C" void kernel_launch(void* const* d_in, const int* in_sizes, int n_in,
                              void* d_out, int out_size, void* d_ws, size_t ws_size,
                              hipStream_t stream) {
  const int*   inp  = (const int*)d_in[0];
  const int*   mask = (const int*)d_in[1];
  const float* emb  = (const float*)d_in[2];
  const float* pos  = (const float*)d_in[3];
  const float* Wq   = (const float*)d_in[4];
  const float* bq   = (const float*)d_in[5];
  const float* Wk   = (const float*)d_in[6];
  const float* bk   = (const float*)d_in[7];
  const float* Wv   = (const float*)d_in[8];
  const float* bv   = (const float*)d_in[9];
  const float* Wo   = (const float*)d_in[10];
  const float* bo   = (const float*)d_in[11];
  const float* ln1g = (const float*)d_in[12];
  const float* ln1b = (const float*)d_in[13];
  const float* W1   = (const float*)d_in[14];
  const float* b1   = (const float*)d_in[15];
  const float* W2   = (const float*)d_in[16];
  const float* b2   = (const float*)d_in[17];
  const float* ln2g = (const float*)d_in[18];
  const float* ln2b = (const float*)d_in[19];
  float* out = (float*)d_out;

  // workspace carve-up
  size_t off = 0;
  auto carve = [&](size_t bytes) -> char* {
    char* p = (char*)d_ws + off;
    off += (bytes + 255) & ~(size_t)255;
    return p;
  };
  float* x    = (float*)carve((size_t)N_TOK * D_MODEL * 4);
  float* tmp  = (float*)carve((size_t)N_TOK * D_MODEL * 4);
  float* x1   = (float*)carve((size_t)N_TOK * D_MODEL * 4);
  unsigned short* xb  = (unsigned short*)carve((size_t)N_TOK * D_MODEL * 2);
  unsigned short* x1b = (unsigned short*)carve((size_t)N_TOK * D_MODEL * 2);
  unsigned short* qbf = (unsigned short*)carve((size_t)N_TOK * D_MODEL * 2);
  unsigned short* kbf = (unsigned short*)carve((size_t)N_TOK * D_MODEL * 2);
  unsigned short* vbf = (unsigned short*)carve((size_t)N_TOK * D_MODEL * 2);
  unsigned short* obf = (unsigned short*)carve((size_t)N_TOK * D_MODEL * 2);
  unsigned short* hbf = (unsigned short*)carve((size_t)N_TOK * FFN_DIM * 2);
  unsigned short* wqb = (unsigned short*)carve((size_t)N_LAYERS * D_MODEL * D_MODEL * 2);
  unsigned short* wkb = (unsigned short*)carve((size_t)N_LAYERS * D_MODEL * D_MODEL * 2);
  unsigned short* wvb = (unsigned short*)carve((size_t)N_LAYERS * D_MODEL * D_MODEL * 2);
  unsigned short* wob = (unsigned short*)carve((size_t)N_LAYERS * D_MODEL * D_MODEL * 2);
  unsigned short* w1b = (unsigned short*)carve((size_t)N_LAYERS * FFN_DIM * D_MODEL * 2);
  unsigned short* w2b = (unsigned short*)carve((size_t)N_LAYERS * D_MODEL * FFN_DIM * 2);
  (void)in_sizes; (void)n_in; (void)out_size; (void)ws_size;

  auto cvt = [&](const float* src, unsigned short* dst, size_t n) {
    int n4 = (int)(n / 4);
    cvt_bf16_kernel<<<(n4 + 255) / 256, 256, 0, stream>>>(src, dst, n4);
  };

  // weights -> bf16 (once per launch)
  cvt(Wq, wqb, (size_t)N_LAYERS * D_MODEL * D_MODEL);
  cvt(Wk, wkb, (size_t)N_LAYERS * D_MODEL * D_MODEL);
  cvt(Wv, wvb, (size_t)N_LAYERS * D_MODEL * D_MODEL);
  cvt(Wo, wob, (size_t)N_LAYERS * D_MODEL * D_MODEL);
  cvt(W1, w1b, (size_t)N_LAYERS * FFN_DIM * D_MODEL);
  cvt(W2, w2b, (size_t)N_LAYERS * D_MODEL * FFN_DIM);

  // x = (emb[input] + pos) * mask
  embed_kernel<<<(N_TOK * 128) / 256, 256, 0, stream>>>(inp, mask, emb, pos, x);

  const dim3 gD(N_TOK / 128, D_MODEL / 128);   // GEMMs with N=512
  const dim3 gF(N_TOK / 128, FFN_DIM / 128);   // GEMM with N=2048
  const dim3 gAtt(S_LEN / 128, B_SZ * N_HEADS);
  const int lnBlocks = N_TOK / 8;

  for (int i = 0; i < N_LAYERS; ++i) {
    const unsigned short* Wq_i = wqb + (size_t)i * D_MODEL * D_MODEL;
    const unsigned short* Wk_i = wkb + (size_t)i * D_MODEL * D_MODEL;
    const unsigned short* Wv_i = wvb + (size_t)i * D_MODEL * D_MODEL;
    const unsigned short* Wo_i = wob + (size_t)i * D_MODEL * D_MODEL;
    const unsigned short* W1_i = w1b + (size_t)i * FFN_DIM * D_MODEL;
    const unsigned short* W2_i = w2b + (size_t)i * D_MODEL * FFN_DIM;

    cvt(x, xb, (size_t)N_TOK * D_MODEL);

    // Q, K, V projections (bf16 out)
    gemm_bt_kernel<false, true><<<gD, 256, 0, stream>>>(
        xb, Wq_i, bq + i * D_MODEL, nullptr, qbf, N_TOK, D_MODEL, D_MODEL);
    gemm_bt_kernel<false, true><<<gD, 256, 0, stream>>>(
        xb, Wk_i, bk + i * D_MODEL, nullptr, kbf, N_TOK, D_MODEL, D_MODEL);
    gemm_bt_kernel<false, true><<<gD, 256, 0, stream>>>(
        xb, Wv_i, bv + i * D_MODEL, nullptr, vbf, N_TOK, D_MODEL, D_MODEL);

    // attention
    flash_attn_kernel<<<gAtt, 256, 0, stream>>>(qbf, kbf, vbf, mask, obf);

    // output projection + residual, then LN1
    gemm_bt_kernel<false, false><<<gD, 256, 0, stream>>>(
        obf, Wo_i, bo + i * D_MODEL, x, tmp, N_TOK, D_MODEL, D_MODEL);
    layernorm_kernel<<<lnBlocks, 256, 0, stream>>>(
        tmp, ln1g + i * D_MODEL, ln1b + i * D_MODEL, nullptr, x1);

    cvt(x1, x1b, (size_t)N_TOK * D_MODEL);

    // FFN1 (relu, bf16 out), FFN2 (+residual), LN2 (* mask)
    gemm_bt_kernel<true, true><<<gF, 256, 0, stream>>>(
        x1b, W1_i, b1 + i * FFN_DIM, nullptr, hbf, N_TOK, FFN_DIM, D_MODEL);
    gemm_bt_kernel<false, false><<<gD, 256, 0, stream>>>(
        hbf, W2_i, b2 + i * D_MODEL, x1, tmp, N_TOK, D_MODEL, FFN_DIM);
    layernorm_kernel<<<lnBlocks, 256, 0, stream>>>(
        tmp, ln2g + i * D_MODEL, ln2b + i * D_MODEL, mask, x);
  }

  pool_kernel<<<B_SZ, 256, 0, stream>>>(x, mask, out);
}
